// Block_63282048139837
// MI455X (gfx1250) — matrix operand order repaired and checked by reading, not compiled, not run on hardware
//
#include <hip/hip_runtime.h>
#include <hip/hip_bf16.h>
#include <math.h>

// ---------------------------------------------------------------------------
// Model constants (from reference): D=1024, 16 heads x 64, T=2048, B=2,
// 8 experts top-2, capacity = ceil(2*4096/8) = 1024.
// ---------------------------------------------------------------------------
#define DMODEL   1024
#define NHEAD    16
#define HDIM     64
#define RHALF    32
#define SEQT     2048
#define BATCH    2
#define NTOK     (BATCH*SEQT)        // 4096
#define NEXP     8
#define CAPACITY 1024

typedef __bf16 v16bf __attribute__((ext_vector_type(16)));
typedef float  v8f   __attribute__((ext_vector_type(8)));
typedef float  v4f   __attribute__((ext_vector_type(4)));

__device__ inline v8f wmma_bf16(v16bf a, v16bf b, v8f c) {
  // D(16x16 f32) = A(16x32 bf16) * B(32x16 bf16) + C
  return __builtin_amdgcn_wmma_f32_16x16x32_bf16(false, a, false, b,
                                                 (short)0, c, false, false);
}

// ---------------------------------------------------------------------------
// LayerNorm: one 256-thread block per row of 1024.
// ---------------------------------------------------------------------------
__global__ __launch_bounds__(256) void ln_kernel(const float* __restrict__ x,
                                                 const float* __restrict__ g,
                                                 const float* __restrict__ b,
                                                 float* __restrict__ out) {
  const int row = blockIdx.x;
  const int tid = threadIdx.x;
  const float* xr = x + (size_t)row * DMODEL;
  __shared__ float red[256];

  float vals[4];
  float s = 0.f;
  for (int i = 0; i < 4; ++i) { vals[i] = xr[tid + i * 256]; s += vals[i]; }
  red[tid] = s;
  __syncthreads();
  for (int off = 128; off > 0; off >>= 1) {
    if (tid < off) red[tid] += red[tid + off];
    __syncthreads();
  }
  const float mu = red[0] * (1.f / DMODEL);
  __syncthreads();

  float vs = 0.f;
  for (int i = 0; i < 4; ++i) { float d = vals[i] - mu; vs += d * d; }
  red[tid] = vs;
  __syncthreads();
  for (int off = 128; off > 0; off >>= 1) {
    if (tid < off) red[tid] += red[tid + off];
    __syncthreads();
  }
  const float inv = rsqrtf(red[0] * (1.f / DMODEL) + 1e-5f);

  for (int i = 0; i < 4; ++i) {
    int c = tid + i * 256;
    out[(size_t)row * DMODEL + c] = (vals[i] - mu) * inv * g[c] + b[c];
  }
}

// ---------------------------------------------------------------------------
// Generic bf16 WMMA GEMM: C[M,N] = A[M,K] * B[K,N], fp32 in memory,
// converted to bf16 at fragment-load time, b128 global loads throughout.
// Block = 8 waves, tile 256x64; wave owns 32x64 (2x4 WMMA accumulators).
// B tiles are double-buffered in LDS: while computing from buffer `cur`,
// the block stages the next 32x64 tile into `cur^1`.
// Epilogues: 0 plain, 1 residual + dual store, 2 bias+GELU, 3 bias+scatter.
// ---------------------------------------------------------------------------
template <int EPI>
__global__ __launch_bounds__(256) void gemm_kernel(
    const float* __restrict__ A, const float* __restrict__ B,
    float* __restrict__ C, float* __restrict__ C2,
    const float* __restrict__ res, const float* __restrict__ bias,
    const int* __restrict__ tok, const float* __restrict__ tw,
    float* __restrict__ scat, int M, int N, int K) {
  __shared__ __bf16 ldsB[2][64 * 32];           // [buf][n_local][k_local]
  const int tid = threadIdx.x;
  const int lane = tid & 31;
  const int w = tid >> 5;
  const int lrow = lane & 15;
  const int lhalf = lane >> 4;
  const int n0 = blockIdx.x * 64;
  const int m0 = blockIdx.y * 256 + w * 32;

  const v4f* __restrict__ B4 = reinterpret_cast<const v4f*>(B);
  const size_t ldb4 = (size_t)(N >> 2);

  // Stage one 32x64 B tile (f32 -> bf16, transposed [n][k]) into buffer.
  auto stage = [&](int kko, __bf16* bufp) {
    for (int i = tid; i < 512; i += 256) {      // 512 float4s = 2048 elements
      const int kI = i >> 4;                    // 0..31
      const int n4 = i & 15;                    // group of 4 columns
      v4f bv = B4[(size_t)(kko + kI) * ldb4 + (n0 >> 2) + n4];
      bufp[(n4 * 4 + 0) * 32 + kI] = (__bf16)bv[0];
      bufp[(n4 * 4 + 1) * 32 + kI] = (__bf16)bv[1];
      bufp[(n4 * 4 + 2) * 32 + kI] = (__bf16)bv[2];
      bufp[(n4 * 4 + 3) * 32 + kI] = (__bf16)bv[3];
    }
  };

  v8f acc[2][4];
  for (int s = 0; s < 2; ++s)
    for (int t = 0; t < 4; ++t)
      for (int r = 0; r < 8; ++r) acc[s][t][r] = 0.f;

  stage(0, ldsB[0]);
  for (int kk = 0; kk < K; kk += 32) {
    const int cur = (kk >> 5) & 1;
    __syncthreads();                            // staging of `cur` visible
    if (kk + 32 < K) stage(kk + 32, ldsB[cur ^ 1]);

    // A fragments (16x32), ISA 16-bit A layout: lane holds row lane&15,
    // elements 0..7 -> K = koff..+7, 8..15 -> K = 16+koff.., koff=8*(l>>4).
    v16bf af[2];
    for (int s = 0; s < 2; ++s) {
      const v4f* ap4 = reinterpret_cast<const v4f*>(
          A + (size_t)(m0 + s * 16 + lrow) * K + kk + (lhalf << 3));
      const v4f a0 = ap4[0], a1 = ap4[1];       // K = koff .. koff+7
      const v4f a2 = ap4[4], a3 = ap4[5];       // K = 16+koff .. 16+koff+7
      for (int j = 0; j < 4; ++j) {
        af[s][j]      = (__bf16)a0[j];
        af[s][4 + j]  = (__bf16)a1[j];
        af[s][8 + j]  = (__bf16)a2[j];
        af[s][12 + j] = (__bf16)a3[j];
      }
    }
    for (int t = 0; t < 4; ++t) {
      // B fragment (32x16): lane holds col lane&15, K = 8*(l>>4)+j (j<8), 16+8*(l>>4)+j-8 (j>=8)
      v16bf bfrag;
      const __bf16* bp = &ldsB[cur][(t * 16 + lrow) * 32 + (lhalf << 3)];
      for (int j = 0; j < 16; ++j) bfrag[j] = bp[(j & 7) + ((j >> 3) << 4)];
      acc[0][t] = wmma_bf16(af[0], bfrag, acc[0][t]);
      acc[1][t] = wmma_bf16(af[1], bfrag, acc[1][t]);
    }
  }

  // Epilogue. C layout: VGPR r -> row r+8*(l>>4), col = lane&15.
  for (int s = 0; s < 2; ++s) {
    for (int t = 0; t < 4; ++t) {
      for (int r = 0; r < 8; ++r) {
        const int m = m0 + s * 16 + r + (lhalf << 3);
        const int n = n0 + t * 16 + lrow;
        float v = acc[s][t][r];
        const size_t off = (size_t)m * N + n;
        if (EPI == 0) {
          C[off] = v;
        } else if (EPI == 1) {
          float y = v + res[off];
          C[off] = y;
          C2[off] = y;
        } else if (EPI == 2) {
          float y = v + bias[n];
          C[off] = 0.5f * y * (1.f + erff(y * 0.7071067811865475f));
        } else {
          float y = v + bias[n];
          int tkn = tok[m];
          if (tkn >= 0) atomicAdd(scat + (size_t)tkn * DMODEL + n, y * tw[m]);
        }
      }
    }
  }
}

// ---------------------------------------------------------------------------
// RoPE + split qkv; also emits k (roped) and v into d_out regions.
// One thread per (token, head, i<32).
// ---------------------------------------------------------------------------
__global__ __launch_bounds__(256) void rope_kernel(
    const float* __restrict__ qkv, float* __restrict__ qout,
    float* __restrict__ kws, float* __restrict__ vws,
    float* __restrict__ kout, float* __restrict__ vout) {
  const int idx = blockIdx.x * 256 + threadIdx.x;   // NTOK*NHEAD*RHALF
  const int i = idx & 31;
  const int h = (idx >> 5) & 15;
  const int n = idx >> 9;          // flat token
  const int t = n & (SEQT - 1);    // position
  const size_t base = (size_t)n * (3 * DMODEL) + h * (3 * HDIM);
  const float q1 = qkv[base + i],        q2 = qkv[base + 32 + i];
  const float k1 = qkv[base + 64 + i],   k2 = qkv[base + 96 + i];
  const float v1 = qkv[base + 128 + i],  v2 = qkv[base + 160 + i];
  const float inv_freq = powf(10000.f, -(float)(2 * i) * (1.f / 64.f));
  const float ang = (float)t * inv_freq;
  float sv, cv;
  sincosf(ang, &sv, &cv);
  const size_t ob = (size_t)n * DMODEL + h * HDIM + i;
  qout[ob] = q1 * cv - q2 * sv;
  qout[ob + 32] = q1 * sv + q2 * cv;
  const float kr1 = k1 * cv - k2 * sv, kr2 = k1 * sv + k2 * cv;
  kws[ob] = kr1;  kws[ob + 32] = kr2;
  kout[ob] = kr1; kout[ob + 32] = kr2;
  vws[ob] = v1;   vws[ob + 32] = v2;
  vout[ob] = v1;  vout[ob + 32] = v2;
}

// ---------------------------------------------------------------------------
// Flash attention: one wave per (b, h, 16-row q tile). Online softmax,
// scores via 2 WMMAs/16-key tile (d split 2x32), P*V via 4 WMMAs/32-key
// block. S/P converted between C-layout and A-layout via per-block LDS.
// Q/K fragment loads are b128 (addresses 16B-aligned by construction).
// ---------------------------------------------------------------------------
__global__ __launch_bounds__(32) void attn_kernel(const float* __restrict__ q,
                                                  const float* __restrict__ k,
                                                  const float* __restrict__ v,
                                                  float* __restrict__ o) {
  const int blk = blockIdx.x;           // ((b*NHEAD + h) * (SEQT/16)) + qt
  const int qt = blk & 127;
  const int bh = blk >> 7;
  const int h = bh & 15;
  const int b = bh >> 4;
  const int lane = threadIdx.x;
  const int lrow = lane & 15;
  const int lhalf = lane >> 4;
  const int qbase = qt * 16;

  const float* qb = q + (size_t)b * SEQT * DMODEL + h * HDIM;
  const float* kb = k + (size_t)b * SEQT * DMODEL + h * HDIM;
  const float* vb = v + (size_t)b * SEQT * DMODEL + h * HDIM;

  __shared__ float  ldsS[16 * 32];
  __shared__ __bf16 ldsP[16 * 32];

  // Q fragments, pre-scaled by 1/sqrt(64).
  v16bf qf[2];
  {
    const int row = qbase + lrow;
    for (int half = 0; half < 2; ++half) {
      const v4f* qp4 = reinterpret_cast<const v4f*>(
          qb + (size_t)row * DMODEL + half * 32 + (lhalf << 4));
      const v4f a0 = qp4[0], a1 = qp4[1], a2 = qp4[2], a3 = qp4[3];
      for (int j = 0; j < 4; ++j) {
        qf[half][j]      = (__bf16)(a0[j] * 0.125f);
        qf[half][4 + j]  = (__bf16)(a1[j] * 0.125f);
        qf[half][8 + j]  = (__bf16)(a2[j] * 0.125f);
        qf[half][12 + j] = (__bf16)(a3[j] * 0.125f);
      }
    }
  }

  v8f ofrag[4];
  for (int t = 0; t < 4; ++t)
    for (int r = 0; r < 8; ++r) ofrag[t][r] = 0.f;
  float m_run = -3.0e38f, l_run = 0.f;

  const int qend = qbase + 15;
  const int nblk = (qend >> 5) + 1;
  for (int jb = 0; jb < nblk; ++jb) {
    const int j0 = jb << 5;
    v8f s0, s1;
    for (int r = 0; r < 8; ++r) { s0[r] = 0.f; s1[r] = 0.f; }
    for (int half = 0; half < 2; ++half) {
      v16bf kf;
      const v4f* kp4 = reinterpret_cast<const v4f*>(
          kb + (size_t)(j0 + lrow) * DMODEL + half * 32 + (lhalf << 4));
      for (int c = 0; c < 4; ++c) {
        const v4f kv = kp4[c];
        for (int j = 0; j < 4; ++j) kf[c * 4 + j] = (__bf16)kv[j];
      }
      s0 = wmma_bf16(qf[half], kf, s0);
      const v4f* kp4b = reinterpret_cast<const v4f*>(
          kb + (size_t)(j0 + 16 + lrow) * DMODEL + half * 32 + (lhalf << 4));
      for (int c = 0; c < 4; ++c) {
        const v4f kv = kp4b[c];
        for (int j = 0; j < 4; ++j) kf[c * 4 + j] = (__bf16)kv[j];
      }
      s1 = wmma_bf16(qf[half], kf, s1);
    }
    // Causal mask + dump S (C layout) to LDS.
    for (int r = 0; r < 8; ++r) {
      const int mrow = qbase + r + (lhalf << 3);
      const int key0 = j0 + lrow;
      ldsS[(r + (lhalf << 3)) * 32 + lrow]      = (key0 <= mrow)      ? s0[r] : -3.0e38f;
      ldsS[(r + (lhalf << 3)) * 32 + 16 + lrow] = (key0 + 16 <= mrow) ? s1[r] : -3.0e38f;
    }
    __syncthreads();

    // Online softmax: lane handles (row = lane&15, half of 32 keys).
    float sv[16];
    float lm = -3.0e38f;
    for (int j = 0; j < 16; ++j) {
      sv[j] = ldsS[lrow * 32 + (lhalf << 4) + j];
      lm = fmaxf(lm, sv[j]);
    }
    lm = fmaxf(lm, __shfl_xor(lm, 16, 32));
    const float mnew = fmaxf(m_run, lm);
    const float alpha = __expf(m_run - mnew);
    float ps = 0.f;
    for (int j = 0; j < 16; ++j) {
      float p = __expf(sv[j] - mnew);
      ps += p;
      ldsP[lrow * 32 + (lhalf << 4) + j] = (__bf16)p;
    }
    ps += __shfl_xor(ps, 16, 32);
    l_run = l_run * alpha + ps;
    m_run = mnew;

    // Rescale O accumulators by per-row alpha (row r+8*lhalf lives on lane==row).
    for (int r = 0; r < 8; ++r) {
      const float a = __shfl(alpha, r + (lhalf << 3), 32);
      for (int t = 0; t < 4; ++t) ofrag[t][r] *= a;
    }
    __syncthreads();

    // P A-fragment from LDS, V B-fragments from global, accumulate O.
    v16bf pf;
    for (int j = 0; j < 8; ++j) pf[j]     = ldsP[lrow * 32 + (lhalf << 4) + j];
    for (int j = 0; j < 8; ++j) pf[8 + j] = ldsP[lrow * 32 + 8 + (lhalf << 4) + j];
    for (int t = 0; t < 4; ++t) {
      v16bf vf;
      const int d = t * 16 + lrow;
      const int keyb = j0 + (lhalf << 4);
      for (int j = 0; j < 16; ++j)
        vf[j] = (__bf16)vb[(size_t)(keyb + j) * DMODEL + d];
      ofrag[t] = wmma_bf16(pf, vf, ofrag[t]);
    }
    __syncthreads();
  }

  // Normalize and store.
  for (int r = 0; r < 8; ++r) {
    const int rowi = r + (lhalf << 3);
    const float li = __shfl(l_run, rowi, 32);
    const float inv = 1.f / li;
    const int qrow = qbase + rowi;
    for (int t = 0; t < 4; ++t) {
      const int d = t * 16 + lrow;
      o[((size_t)b * SEQT + qrow) * DMODEL + h * HDIM + d] = ofrag[t][r] * inv;
    }
  }
}

// ---------------------------------------------------------------------------
// Gating: one wave per token. 8-way dot over D=1024, softmax, top-2.
// ---------------------------------------------------------------------------
__global__ __launch_bounds__(32) void gate_kernel(const float* __restrict__ ffin,
                                                  const float* __restrict__ Wg,
                                                  int* __restrict__ gidx,
                                                  float* __restrict__ gval) {
  const int tkn = blockIdx.x;
  const int lane = threadIdx.x;
  float acc[NEXP];
  for (int e = 0; e < NEXP; ++e) acc[e] = 0.f;
  const float* xr = ffin + (size_t)tkn * DMODEL;
  for (int d = lane; d < DMODEL; d += 32) {
    const float xv = xr[d];
    const float* wr = Wg + (size_t)d * NEXP;
    for (int e = 0; e < NEXP; ++e) acc[e] += xv * wr[e];
  }
  for (int off = 16; off > 0; off >>= 1)
    for (int e = 0; e < NEXP; ++e) acc[e] += __shfl_xor(acc[e], off, 32);
  if (lane == 0) {
    float mx = acc[0];
    for (int e = 1; e < NEXP; ++e) mx = fmaxf(mx, acc[e]);
    float g[NEXP], sum = 0.f;
    for (int e = 0; e < NEXP; ++e) { g[e] = __expf(acc[e] - mx); sum += g[e]; }
    const float inv = 1.f / sum;
    for (int e = 0; e < NEXP; ++e) g[e] *= inv;
    int i0 = 0; float v0 = g[0];
    for (int e = 1; e < NEXP; ++e) if (g[e] > v0) { v0 = g[e]; i0 = e; }
    int i1 = (i0 == 0) ? 1 : 0; float v1 = g[i1];
    for (int e = 0; e < NEXP; ++e)
      if (e != i0 && g[e] > v1) { v1 = g[e]; i1 = e; }
    gidx[tkn * 2] = i0; gidx[tkn * 2 + 1] = i1;
    gval[tkn * 2] = v0; gval[tkn * 2 + 1] = v1;
  }
}

// ---------------------------------------------------------------------------
// Deterministic capacity routing: one wave per expert scans all 8192
// (token,k) assignments in order; ballot prefix gives ordered slots.
// ---------------------------------------------------------------------------
__global__ __launch_bounds__(32) void route_kernel(const int* __restrict__ gidx,
                                                   const float* __restrict__ gval,
                                                   int* __restrict__ stok,
                                                   float* __restrict__ sw) {
  const int e = blockIdx.x;
  const int lane = threadIdx.x;
  int count = 0;
  for (int base = 0; base < NTOK * 2; base += 32) {
    const int i = base + lane;
    const bool f = (gidx[i] == e);
    const unsigned long long bal = __ballot(f);
    const int pre = __popcll(bal & ((1ull << lane) - 1ull));
    if (f) {
      const int s = count + pre;
      if (s < CAPACITY) {
        stok[e * CAPACITY + s] = i >> 1;
        sw[e * CAPACITY + s] = gval[i];
      }
    }
    count += __popcll(bal);
  }
  for (int s = count + lane; s < CAPACITY; s += 32) {
    stok[e * CAPACITY + s] = -1;
    sw[e * CAPACITY + s] = 0.f;
  }
}

__global__ __launch_bounds__(256) void gather_kernel(const float* __restrict__ ffin,
                                                     const int* __restrict__ stok,
                                                     float* __restrict__ xe) {
  const int idx = blockIdx.x * 256 + threadIdx.x;  // NEXP*CAPACITY*DMODEL
  const int row = idx >> 10;
  const int d = idx & (DMODEL - 1);
  const int tkn = stok[row];
  xe[idx] = (tkn >= 0) ? ffin[(size_t)tkn * DMODEL + d] : 0.f;
}

__global__ void aux_kernel(float* p) { *p = 0.f; }

// ---------------------------------------------------------------------------
// Host launcher. Inputs (setup_inputs order):
// 0 x, 1 ln1_g, 2 ln1_b, 3 ln2_g, 4 ln2_b, 5 Wqkv, 6 Wproj, 7 Wgate,
// 8 W1, 9 b1, 10 W2, 11 b2.   Output: x | aux | k | v (flat f32).
// ---------------------------------------------------------------------------
extern "C" void kernel_launch(void* const* d_in, const int* in_sizes, int n_in,
                              void* d_out, int out_size, void* d_ws, size_t ws_size,
                              hipStream_t stream) {
  (void)in_sizes; (void)n_in; (void)out_size; (void)ws_size;
  const float* x     = (const float*)d_in[0];
  const float* ln1_g = (const float*)d_in[1];
  const float* ln1_b = (const float*)d_in[2];
  const float* ln2_g = (const float*)d_in[3];
  const float* ln2_b = (const float*)d_in[4];
  const float* Wqkv  = (const float*)d_in[5];
  const float* Wproj = (const float*)d_in[6];
  const float* Wgate = (const float*)d_in[7];
  const float* W1    = (const float*)d_in[8];
  const float* b1    = (const float*)d_in[9];
  const float* W2    = (const float*)d_in[10];
  const float* b2    = (const float*)d_in[11];

  float* out = (float*)d_out;
  const size_t XN = (size_t)NTOK * DMODEL;          // 4,194,304
  float* xout = out;                                // [NTOK, D]
  float* aux  = out + XN;                           // [1]
  float* kout = out + XN + 1;                       // [NTOK, D] (roped k)
  float* vout = out + XN + 1 + XN;                  // [NTOK, D]

  // Workspace layout (lifetimes overlapped):
  float* ws = (float*)d_ws;
  float* wsA = ws;                 // 4M floats : xnorm1, later attn out
  float* wsB = wsA + XN;           // 12M floats: qkv, later x1|ffin|h
  float* wsC = wsB + 3 * XN;       // 4M floats : q_rope, later xe (low half)
  float* wsD = wsC + XN;           // 4M floats : k_rope, later xe (high half)
  float* wsE = wsD + XN;           // 4M floats : v (aligned copy for attn)
  float* gvalp = wsE + XN;         // 8192
  float* swp   = gvalp + NTOK * 2; // 8192
  int*   gidxp = (int*)(swp + NEXP * CAPACITY);   // 8192 ints
  int*   stokp = gidxp + NTOK * 2;                // 8192 ints

  float* xnorm1 = wsA;
  float* qkv    = wsB;
  float* qrope  = wsC;
  float* krope  = wsD;
  float* vcopy  = wsE;
  float* attno  = wsA;             // xnorm1 dead after QKV GEMM
  float* x1     = wsB;             // qkv dead after rope+attn
  float* ffin   = wsB + XN;
  float* hbuf   = wsB + 2 * XN;
  float* xe     = wsC;             // spans wsC+wsD (8M floats), q/k dead

  // 1) LN1
  ln_kernel<<<NTOK, 256, 0, stream>>>(x, ln1_g, ln1_b, xnorm1);
  // 2) QKV GEMM: [4096,1024] x [1024,3072]
  gemm_kernel<0><<<dim3(3 * DMODEL / 64, NTOK / 256), 256, 0, stream>>>(
      xnorm1, Wqkv, qkv, nullptr, nullptr, nullptr, nullptr, nullptr, nullptr,
      NTOK, 3 * DMODEL, DMODEL);
  // 3) RoPE + emit k,v outputs
  rope_kernel<<<(NTOK * NHEAD * RHALF) / 256, 256, 0, stream>>>(
      qkv, qrope, krope, vcopy, kout, vout);
  // 4) Flash attention (1 wave per 16-row q tile)
  attn_kernel<<<BATCH * NHEAD * (SEQT / 16), 32, 0, stream>>>(qrope, krope, vcopy, attno);
  // 5) Output projection + residual; also populate d_out x region
  gemm_kernel<1><<<dim3(DMODEL / 64, NTOK / 256), 256, 0, stream>>>(
      attno, Wproj, x1, xout, x, nullptr, nullptr, nullptr, nullptr,
      NTOK, DMODEL, DMODEL);
  // 6) LN2
  ln_kernel<<<NTOK, 256, 0, stream>>>(x1, ln2_g, ln2_b, ffin);
  // 7) Gating + routing + gather
  gate_kernel<<<NTOK, 32, 0, stream>>>(ffin, Wgate, gidxp, gvalp);
  route_kernel<<<NEXP, 32, 0, stream>>>(gidxp, gvalp, stokp, swp);
  gather_kernel<<<(NEXP * CAPACITY * DMODEL) / 256, 256, 0, stream>>>(ffin, stokp, xe);
  aux_kernel<<<1, 1, 0, stream>>>(aux);
  // 8) Expert FFNs (sequential; reuse h buffer), scatter-add into xout
  for (int e = 0; e < NEXP; ++e) {
    gemm_kernel<2><<<dim3(4 * DMODEL / 64, CAPACITY / 256), 256, 0, stream>>>(
        xe + (size_t)e * CAPACITY * DMODEL, W1 + (size_t)e * DMODEL * 4 * DMODEL,
        hbuf, nullptr, nullptr, b1 + (size_t)e * 4 * DMODEL,
        nullptr, nullptr, nullptr, CAPACITY, 4 * DMODEL, DMODEL);
    gemm_kernel<3><<<dim3(DMODEL / 64, CAPACITY / 256), 256, 0, stream>>>(
        hbuf, W2 + (size_t)e * 4 * DMODEL * DMODEL,
        nullptr, nullptr, nullptr, b2 + (size_t)e * DMODEL,
        stokp + e * CAPACITY, swp + e * CAPACITY, xout,
        CAPACITY, DMODEL, 4 * DMODEL);
  }
}
